// RNNAttention_21758304321700
// MI455X (gfx1250) — matrix operand order repaired and checked by reading, not compiled, not run on hardware
//
#include <hip/hip_runtime.h>

// ---------------------------------------------------------------------------
// Problem constants (match reference setup_inputs exactly)
// ---------------------------------------------------------------------------
#define B_    512
#define C_    512
#define HID_  512
#define L_    256
#define L1_   257
#define G_    2048   // 4*hid
#define KCAT_ 1024   // c + hid  (concatenated [ctx | h] GEMM K)
#define T_    26
#define NC_   38

typedef __attribute__((ext_vector_type(16))) __bf16 v16bf;
typedef __attribute__((ext_vector_type(8)))  __bf16 v8bf;
typedef __attribute__((ext_vector_type(8)))  float  v8f;

__device__ __forceinline__ __bf16 f2bf(float f) {
  unsigned u = __builtin_bit_cast(unsigned, f);
  u += 0x7FFFu + ((u >> 16) & 1u);            // round-to-nearest-even
  unsigned short h = (unsigned short)(u >> 16);
  return __builtin_bit_cast(__bf16, h);
}

__device__ __forceinline__ float wave_sum(float s) {
#pragma unroll
  for (int off = 16; off > 0; off >>= 1) s += __shfl_xor(s, off, 32);
  return s;
}

// gfx1250 hardware tanh (TRANS op, co-executes with VALU); fallback to libm.
__device__ __forceinline__ float fast_tanh(float x) {
#if __has_builtin(__builtin_amdgcn_tanhf)
  return __builtin_amdgcn_tanhf(x);
#elif __has_builtin(__builtin_amdgcn_tanh_f32)
  return __builtin_amdgcn_tanh_f32(x);
#else
  return tanhf(x);
#endif
}

__device__ __forceinline__ float fast_sigmoid(float x) {
  return 0.5f * fast_tanh(0.5f * x) + 0.5f;
}

__device__ __forceinline__ void wait_async0() {
#if __has_builtin(__builtin_amdgcn_s_wait_asynccnt)
  __builtin_amdgcn_s_wait_asynccnt(0);
#else
  asm volatile("s_wait_asynccnt 0" ::: "memory");
#endif
}

// ---------------------------------------------------------------------------
// Generic bf16 NT GEMM:  C[M,N] = A[M,K] (row-major, stride lda) x B[N,K]^T + bias
// Requires M % 64 == 0, N % 64 == 0, K % 32 == 0 (true for all uses here).
// Block = 128 threads (4 waves); each wave -> 16x64 tile via 4 WMMAs / K-chunk.
// B tile (64 cols x 32 K, 4 KB) is staged into double-buffered LDS with
// GLOBAL_LOAD_ASYNC_TO_LDS_B128 (ASYNCcnt) and shared by all 4 waves,
// removing the 4x redundant B reads. A streams straight from global/L2.
// A/B fragments use contiguous K: lane half h holds K = 16h..16h+15 in both operands.
// ---------------------------------------------------------------------------
__global__ void __launch_bounds__(128)
gemm_bf16_nt(const __bf16* __restrict__ A, int lda,
             const __bf16* __restrict__ Bm,          // [N,K] row-major, ldb == K
             const float*  __restrict__ bias,        // [N] or nullptr
             float*  __restrict__ Cf,                // f32 out or nullptr
             __bf16* __restrict__ Cbf,               // bf16 out or nullptr
             int M, int N, int K) {
  __shared__ __align__(32) __bf16 bs[2][64 * 32];    // 2 x 4 KB B tiles
  const int tid  = threadIdx.x;
  const int wave = tid >> 5;
  const int lane = tid & 31;
  const int m0   = (blockIdx.x * 4 + wave) * 16;
  const int n0   = blockIdx.y * 64;
  const int mr   = m0 + (lane & 15);
  const int ksel = lane >> 4;                   // lane half selects K sub-block
  const int ncol = lane & 15;

  // Async-copy mapping: 256 x 16B segments per chunk, 2 per thread.
  const int s0 = 2 * tid, s1 = 2 * tid + 1;
  const int r0 = s0 >> 2, q0 = s0 & 3;
  const int r1 = s1 >> 2, q1 = s1 & 3;

  auto issue_chunk = [&](int kc, int buf) {
    const __bf16* g0 = Bm + (size_t)(n0 + r0) * K + kc + q0 * 8;
    const __bf16* g1 = Bm + (size_t)(n0 + r1) * K + kc + q1 * 8;
    unsigned l0 = (unsigned)(uintptr_t)(&bs[buf][r0 * 32 + q0 * 8]);
    unsigned l1 = (unsigned)(uintptr_t)(&bs[buf][r1 * 32 + q1 * 8]);
    asm volatile("global_load_async_to_lds_b128 %0, %1, off"
                 :: "v"(l0), "v"((unsigned long long)(uintptr_t)g0) : "memory");
    asm volatile("global_load_async_to_lds_b128 %0, %1, off"
                 :: "v"(l1), "v"((unsigned long long)(uintptr_t)g1) : "memory");
  };

  v8f acc0 = {}, acc1 = {}, acc2 = {}, acc3 = {};
  const __bf16* arow = A + (size_t)mr * lda + ksel * 16;

  issue_chunk(0, 0);
  wait_async0();
  __syncthreads();

  int buf = 0;
  for (int k = 0; k < K; k += 32) {
    if (k + 32 < K) issue_chunk(k + 32, buf ^ 1);

    // A fragment: per-lane K = ksel*16 + [0..15]
    v8bf alo = *(const v8bf*)(arow + k);
    v8bf ahi = *(const v8bf*)(arow + k + 8);
    v16bf a = __builtin_shufflevector(alo, ahi, 0, 1, 2, 3, 4, 5, 6, 7,
                                      8, 9, 10, 11, 12, 13, 14, 15);
    // B fragments from LDS: lane = column, 16 consecutive K at ksel*16
    const __bf16* bb = &bs[buf][0];
    v16bf f0 = *(const v16bf*)(bb + (ncol +  0) * 32 + ksel * 16);
    v16bf f1 = *(const v16bf*)(bb + (ncol + 16) * 32 + ksel * 16);
    v16bf f2 = *(const v16bf*)(bb + (ncol + 32) * 32 + ksel * 16);
    v16bf f3 = *(const v16bf*)(bb + (ncol + 48) * 32 + ksel * 16);
    acc0 = __builtin_amdgcn_wmma_f32_16x16x32_bf16(false, a, false, f0, (short)0, acc0, false, false);
    acc1 = __builtin_amdgcn_wmma_f32_16x16x32_bf16(false, a, false, f1, (short)0, acc1, false, false);
    acc2 = __builtin_amdgcn_wmma_f32_16x16x32_bf16(false, a, false, f2, (short)0, acc2, false, false);
    acc3 = __builtin_amdgcn_wmma_f32_16x16x32_bf16(false, a, false, f3, (short)0, acc3, false, false);

    wait_async0();      // own async copies into buf^1 complete
    __syncthreads();    // all waves done reading buf / writing buf^1
    buf ^= 1;
  }

  // C/D layout: VGPR r holds row (m0 + ksel*8 + r), column = n-tile + ncol
  v8f accs[4] = {acc0, acc1, acc2, acc3};
  const int mrow = m0 + ksel * 8;
#pragma unroll
  for (int j = 0; j < 4; ++j) {
    const int ncc = n0 + j * 16 + ncol;
    const float bv = bias ? bias[ncc] : 0.0f;
#pragma unroll
    for (int r = 0; r < 8; ++r) {
      const float v = accs[j][r] + bv;
      const size_t idx = (size_t)(mrow + r) * N + ncc;
      if (Cf)  Cf[idx]  = v;
      if (Cbf) Cbf[idx] = f2bf(v);
    }
  }
}

// ---------------------------------------------------------------------------
// Build Hbf[b, l, c] (bf16) from batch_H[b, c, l] via LDS tile transpose.
// grid (L/32, C/32, B), block (32, 8)
// ---------------------------------------------------------------------------
__global__ void __launch_bounds__(256)
build_H_kernel(const float* __restrict__ bh, __bf16* __restrict__ Hbf) {
  __shared__ float tile[32][33];
  const int b  = blockIdx.z;
  const int l0 = blockIdx.x * 32;
  const int c0 = blockIdx.y * 32;
  const int tx = threadIdx.x, ty = threadIdx.y;
#pragma unroll
  for (int i = 0; i < 4; ++i) {
    const int cc = c0 + ty + i * 8;
    tile[ty + i * 8][tx] = bh[((size_t)b * C_ + cc) * L_ + l0 + tx];
  }
  __syncthreads();
#pragma unroll
  for (int i = 0; i < 4; ++i) {
    const int l = l0 + ty + i * 8;
    Hbf[((size_t)b * L1_ + l) * C_ + c0 + tx] = f2bf(tile[tx][ty + i * 8]);
  }
}

__global__ void __launch_bounds__(256)
eos_row_kernel(const float* __restrict__ eos, __bf16* __restrict__ Hbf) {
  const int i = blockIdx.x * 256 + threadIdx.x;   // over B*C
  if (i >= B_ * C_) return;
  const int b = i >> 9, cc = i & 511;
  Hbf[((size_t)b * L1_ + L_) * C_ + cc] = f2bf(eos[cc]);
}

// ---------------------------------------------------------------------------
// Weight conversion / fusion (once)
// ---------------------------------------------------------------------------
__global__ void __launch_bounds__(256)
cvt_weights_kernel(const float* __restrict__ i2h, const float* __restrict__ h2h,
                   const float* __restrict__ wih, const float* __restrict__ whh,
                   const float* __restrict__ bih, const float* __restrict__ bhh,
                   __bf16* __restrict__ i2h_bf, __bf16* __restrict__ h2h_bf,
                   __bf16* __restrict__ wcat, float* __restrict__ bsum) {
  const int i = blockIdx.x * 256 + threadIdx.x;   // over G_*HID_
  if (i < C_ * HID_) { i2h_bf[i] = f2bf(i2h[i]); h2h_bf[i] = f2bf(h2h[i]); }
  if (i < G_) bsum[i] = bih[i] + bhh[i];
  if (i < G_ * HID_) {
    const int n = i / HID_, k = i % HID_;
    wcat[(size_t)n * KCAT_ + k]        = f2bf(wih[i]);
    wcat[(size_t)n * KCAT_ + HID_ + k] = f2bf(whh[i]);
  }
}

__global__ void __launch_bounds__(256)
init_state_kernel(float* __restrict__ hst, float* __restrict__ cst,
                  __bf16* __restrict__ xcat) {
  const int i = blockIdx.x * 256 + threadIdx.x;   // over B*HID
  if (i >= B_ * HID_) return;
  hst[i] = 0.0f; cst[i] = 0.0f;
  xcat[(size_t)(i >> 9) * KCAT_ + HID_ + (i & 511)] = f2bf(0.0f);
}

// ---------------------------------------------------------------------------
// Attention score:  e[b,l] = sum_h score_w[h] * tanh(Hp[b,l,h] + q[b,h]),
// masked positions -> -inf.   grid (ceil(L1/8), B), block 256 (8 waves).
// ---------------------------------------------------------------------------
__global__ void __launch_bounds__(256)
score_kernel(const __bf16* __restrict__ Hp, const float* __restrict__ q,
             const float* __restrict__ sw, const float* __restrict__ mask,
             float* __restrict__ e) {
  __shared__ float qs[HID_];
  __shared__ float sws[HID_];
  const int b = blockIdx.y;
  for (int i = threadIdx.x; i < HID_; i += 256) {
    qs[i]  = q[(size_t)b * HID_ + i];
    sws[i] = sw[i];
  }
  __syncthreads();
  const int wave = threadIdx.x >> 5, lane = threadIdx.x & 31;
  const int l = blockIdx.x * 8 + wave;
  if (l >= L1_) return;
  const bool masked = (l < L_) && ((1.0f - mask[(size_t)b * L_ + l]) >= 0.5f);
  const __bf16* hp = Hp + ((size_t)b * L1_ + l) * HID_;
  float s = 0.0f;
#pragma unroll
  for (int it = 0; it < HID_ / 32; ++it) {
    const int h = it * 32 + lane;
    s += sws[h] * fast_tanh((float)hp[h] + qs[h]);
  }
  s = wave_sum(s);
  if (lane == 0)
    e[(size_t)b * L1_ + l] = masked ? -__builtin_inff() : s;
}

// ---------------------------------------------------------------------------
// Softmax over L1 keys per batch row; writes alpha and char_maps[b,t,:].
// grid (B), block 256.
// ---------------------------------------------------------------------------
__global__ void __launch_bounds__(256)
softmax_kernel(const float* __restrict__ e, float* __restrict__ alpha,
               float* __restrict__ charmaps, int t) {
  __shared__ float redm[8];
  __shared__ float reds[8];
  __shared__ float sa[L1_];
  const int b = blockIdx.x;
  const float* eb = e + (size_t)b * L1_;
  const int wave = threadIdx.x >> 5, lane = threadIdx.x & 31;

  float m = -__builtin_inff();
  for (int l = threadIdx.x; l < L1_; l += 256) m = fmaxf(m, eb[l]);
#pragma unroll
  for (int off = 16; off > 0; off >>= 1) m = fmaxf(m, __shfl_xor(m, off, 32));
  if (lane == 0) redm[wave] = m;
  __syncthreads();
  float mx = redm[0];
#pragma unroll
  for (int i = 1; i < 8; ++i) mx = fmaxf(mx, redm[i]);

  float ssum = 0.0f;
  for (int l = threadIdx.x; l < L1_; l += 256) {
    const float v = __expf(eb[l] - mx);
    sa[l] = v;
    ssum += v;
  }
  ssum = wave_sum(ssum);
  if (lane == 0) reds[wave] = ssum;
  __syncthreads();
  float tot = 0.0f;
#pragma unroll
  for (int i = 0; i < 8; ++i) tot += reds[i];
  const float inv = 1.0f / tot;

  for (int l = threadIdx.x; l < L1_; l += 256) {
    const float a = sa[l] * inv;
    alpha[(size_t)b * L1_ + l] = a;
    charmaps[((size_t)b * T_ + t) * L1_ + l] = a;
  }
}

// ---------------------------------------------------------------------------
// Context:  ctx[b,c] = sum_{l<L} alpha[b,l]*batch_H[b,c,l] + alpha[b,L]*eos[c]
// Written as bf16 into the low half of xcat[b, 0:C].  grid (C/8, B), block 256.
// ---------------------------------------------------------------------------
__global__ void __launch_bounds__(256)
ctx_kernel(const float* __restrict__ bh, const float* __restrict__ eos,
           const float* __restrict__ alpha, __bf16* __restrict__ xcat) {
  __shared__ float sa[L1_];
  const int b = blockIdx.y;
  for (int i = threadIdx.x; i < L1_; i += 256) sa[i] = alpha[(size_t)b * L1_ + i];
  __syncthreads();
  const int wave = threadIdx.x >> 5, lane = threadIdx.x & 31;
  const int cc = blockIdx.x * 8 + wave;
  const float* row = bh + ((size_t)b * C_ + cc) * L_;
  float s = 0.0f;
#pragma unroll
  for (int it = 0; it < L_ / 32; ++it) {
    const int l = it * 32 + lane;
    s += sa[l] * row[l];
  }
  s = wave_sum(s);
  if (lane == 0) {
    s += sa[L_] * eos[cc];
    xcat[(size_t)b * KCAT_ + cc] = f2bf(s);
  }
}

// ---------------------------------------------------------------------------
// LSTM pointwise; writes h (f32 + bf16 into xcat high half) and c state.
// ---------------------------------------------------------------------------
__global__ void __launch_bounds__(256)
lstm_kernel(const float* __restrict__ gates, float* __restrict__ hst,
            float* __restrict__ cst, __bf16* __restrict__ xcat) {
  const int i = blockIdx.x * 256 + threadIdx.x;   // over B*HID
  if (i >= B_ * HID_) return;
  const int b = i >> 9, j = i & 511;
  const float* g = gates + (size_t)b * G_;
  const float ig = fast_sigmoid(g[j]);
  const float fg = fast_sigmoid(g[HID_ + j]);
  const float gg = fast_tanh(g[2 * HID_ + j]);
  const float og = fast_sigmoid(g[3 * HID_ + j]);
  const float cn = fg * cst[i] + ig * gg;
  const float hn = og * fast_tanh(cn);
  cst[i] = cn;
  hst[i] = hn;
  xcat[(size_t)b * KCAT_ + HID_ + j] = f2bf(hn);
}

// ---------------------------------------------------------------------------
// Generator:  logits[b,t,n] = gen_w[n,:] . h[b,:] + gen_b[n].
// grid (ceil(NC/8), B), block 256.
// ---------------------------------------------------------------------------
__global__ void __launch_bounds__(256)
gen_kernel(const float* __restrict__ h, const float* __restrict__ gw,
           const float* __restrict__ gb, float* __restrict__ logits, int t) {
  const int b = blockIdx.y;
  const int wave = threadIdx.x >> 5, lane = threadIdx.x & 31;
  const int n = blockIdx.x * 8 + wave;
  if (n >= NC_) return;
  const float* hr = h  + (size_t)b * HID_;
  const float* wr = gw + (size_t)n * HID_;
  float s = 0.0f;
#pragma unroll
  for (int it = 0; it < HID_ / 32; ++it) {
    const int k = it * 32 + lane;
    s += hr[k] * wr[k];
  }
  s = wave_sum(s);
  if (lane == 0)
    logits[((size_t)b * T_ + t) * NC_ + n] = s + gb[n];
}

// ---------------------------------------------------------------------------
// Host launcher
// ---------------------------------------------------------------------------
extern "C" void kernel_launch(void* const* d_in, const int* in_sizes, int n_in,
                              void* d_out, int out_size, void* d_ws, size_t ws_size,
                              hipStream_t stream) {
  (void)in_sizes; (void)n_in; (void)out_size; (void)ws_size;
  const float* batch_H = (const float*)d_in[0];
  const float* mask    = (const float*)d_in[1];
  const float* i2h_w   = (const float*)d_in[2];
  const float* h2h_w   = (const float*)d_in[3];
  const float* h2h_b   = (const float*)d_in[4];
  const float* score_w = (const float*)d_in[5];
  const float* w_ih    = (const float*)d_in[6];
  const float* w_hh    = (const float*)d_in[7];
  const float* b_ih    = (const float*)d_in[8];
  const float* b_hh    = (const float*)d_in[9];
  const float* gen_w   = (const float*)d_in[10];
  const float* gen_b   = (const float*)d_in[11];
  const float* eos_emb = (const float*)d_in[12];
  // d_in[13] = num_steps (device scalar); fixed to 26 by setup_inputs.

  float* logits   = (float*)d_out;
  float* charmaps = logits + (size_t)B_ * T_ * NC_;

  char* ws = (char*)d_ws;
  auto alloc = [&](size_t bytes) -> char* {
    char* p = ws;
    ws += (bytes + 255) & ~(size_t)255;
    return p;
  };
  const size_t M1 = (size_t)B_ * L1_;                 // 131584 rows (mult of 64)
  __bf16* Hbf    = (__bf16*)alloc(M1 * C_ * 2);       // H, bf16 [b,L1,c]
  __bf16* Hpbf   = (__bf16*)alloc(M1 * HID_ * 2);     // Hp, bf16 (L2-resident set)
  __bf16* i2h_bf = (__bf16*)alloc((size_t)HID_ * C_ * 2);
  __bf16* h2h_bf = (__bf16*)alloc((size_t)HID_ * HID_ * 2);
  __bf16* wcat   = (__bf16*)alloc((size_t)G_ * KCAT_ * 2);
  float*  bsum   = (float*)alloc((size_t)G_ * 4);
  float*  qbuf   = (float*)alloc((size_t)B_ * HID_ * 4);
  float*  ebuf   = (float*)alloc((size_t)B_ * L1_ * 4);
  float*  alpha  = (float*)alloc((size_t)B_ * L1_ * 4);
  float*  gates  = (float*)alloc((size_t)B_ * G_ * 4);
  float*  hst    = (float*)alloc((size_t)B_ * HID_ * 4);
  float*  cst    = (float*)alloc((size_t)B_ * HID_ * 4);
  __bf16* xcat   = (__bf16*)alloc((size_t)B_ * KCAT_ * 2);  // [ctx | h] bf16

  // ---- one-time setup ----
  build_H_kernel<<<dim3(L_ / 32, C_ / 32, B_), dim3(32, 8), 0, stream>>>(batch_H, Hbf);
  eos_row_kernel<<<(B_ * C_ + 255) / 256, 256, 0, stream>>>(eos_emb, Hbf);
  cvt_weights_kernel<<<(G_ * HID_ + 255) / 256, 256, 0, stream>>>(
      i2h_w, h2h_w, w_ih, w_hh, b_ih, b_hh, i2h_bf, h2h_bf, wcat, bsum);
  init_state_kernel<<<(B_ * HID_ + 255) / 256, 256, 0, stream>>>(hst, cst, xcat);

  // Hp[b*L1, hid] = Hbf x i2h^T   (bf16 out -> whole attention set ~134 MB, L2-resident)
  gemm_bf16_nt<<<dim3((unsigned)(M1 / 64), HID_ / 64), 128, 0, stream>>>(
      Hbf, C_, i2h_bf, nullptr, nullptr, Hpbf, (int)M1, HID_, C_);

  // ---- decode loop ----
  for (int t = 0; t < T_; ++t) {
    // q = h x h2h^T + h2h_b   (A = bf16 h slice of xcat, stride KCAT)
    gemm_bf16_nt<<<dim3(B_ / 64, HID_ / 64), 128, 0, stream>>>(
        xcat + HID_, KCAT_, h2h_bf, h2h_b, qbuf, nullptr, B_, HID_, HID_);

    score_kernel<<<dim3((L1_ + 7) / 8, B_), 256, 0, stream>>>(
        Hpbf, qbuf, score_w, mask, ebuf);

    softmax_kernel<<<B_, 256, 0, stream>>>(ebuf, alpha, charmaps, t);

    ctx_kernel<<<dim3(C_ / 8, B_), 256, 0, stream>>>(batch_H, eos_emb, alpha, xcat);

    // gates = [ctx|h] x [w_ih|w_hh]^T + (b_ih + b_hh)   (K = 1024)
    gemm_bf16_nt<<<dim3(B_ / 64, G_ / 64), 128, 0, stream>>>(
        xcat, KCAT_, wcat, bsum, gates, nullptr, B_, G_, KCAT_);

    lstm_kernel<<<(B_ * HID_ + 255) / 256, 256, 0, stream>>>(gates, hst, cst, xcat);

    gen_kernel<<<dim3((NC_ + 7) / 8, B_), 256, 0, stream>>>(
        hst, gen_w, gen_b, logits, t);
  }
}
